// MultiheadAttention_7395933684014
// MI455X (gfx1250) — matrix operand order repaired and checked by reading, not compiled, not run on hardware
//
#include <hip/hip_runtime.h>
#include <hip/hip_bf16.h>

// MI455X / gfx1250: wave32, WMMA bf16 16x16x32, f32 accumulate.

typedef __bf16 bf16_t;
typedef __attribute__((ext_vector_type(16))) __bf16 v16bf;
typedef __attribute__((ext_vector_type(8)))  __bf16 v8bf;
typedef __attribute__((ext_vector_type(4)))  __bf16 v4bf;
typedef __attribute__((ext_vector_type(8)))  float  v8f;
typedef __attribute__((ext_vector_type(4)))  float  v4f;

#define T_LEN 2048
#define BSZ   4
#define EMB   1024
#define NH    16
#define HD    64
#define MROWS (T_LEN * BSZ)   // 8192

__device__ __forceinline__ v8f wmma_bf16(v16bf a, v16bf b, v8f c) {
  // D = A(16x32 bf16) * B(32x16 bf16) + C(16x16 f32)
  return __builtin_amdgcn_wmma_f32_16x16x32_bf16(false, a, false, b, (short)0, c, false, false);
}

// A-fragment (16x32 bf16). Contiguous k order used here (A and B match): lane half = lane>>4;
// element i -> k = k0 + half*16 + i
__device__ __forceinline__ v16bf load_a_frag(const bf16_t* __restrict__ base,
                                             int stride, int row, int k0, int half) {
  const bf16_t* p = base + (size_t)row * stride + k0 + (half << 4);
  v8bf lo = *(const v8bf*)p;          // k0 + half*16 + 0..7
  v8bf hi = *(const v8bf*)(p + 8);    // k0 + half*16 + 8..15
  v16bf r;
#pragma unroll
  for (int i = 0; i < 8; ++i) { r[i] = lo[i]; r[i + 8] = hi[i]; }
  return r;
}

// Same A-fragment pattern but from an LDS-resident bf16 tile (row stride 2048).
__device__ __forceinline__ v16bf load_a_frag_lds(const bf16_t* p) {
  v8bf lo = *(const v8bf*)p;
  v8bf hi = *(const v8bf*)(p + 8);
  v16bf r;
#pragma unroll
  for (int i = 0; i < 8; ++i) { r[i] = lo[i]; r[i + 8] = hi[i]; }
  return r;
}

// ---------------------------------------------------------------------------
// f32 -> bf16 conversion (vectorized x4)
// ---------------------------------------------------------------------------
__global__ void cvt_f32_to_bf16(const float* __restrict__ in,
                                bf16_t* __restrict__ out, int n4) {
  int i = blockIdx.x * blockDim.x + threadIdx.x;
  if (i < n4) {
    v4f v = ((const v4f*)in)[i];
    v4bf o;
#pragma unroll
    for (int j = 0; j < 4; ++j) o[j] = (bf16_t)v[j];
    ((v4bf*)out)[i] = o;
  }
}

// ---------------------------------------------------------------------------
// GEMM: C[m,n] = sum_k A[m,k]*W[n,k] + bias[n]
// A: [8192,1024] bf16 row-major, W: [1024,1024] bf16 row-major (torch Linear W)
// Block = 256 threads = 8 waves (2x4). Block tile 128x128; wave tile 64x32.
// mode 0: f32 out row-major [M,N]            (final projection -> d_out)
// mode 1: Q  -> bf16 [B,H,T,D], scaled 1/8   (fold in 1/sqrt(D))
// mode 2: K  -> bf16 [B,H,T,D]
// mode 3: V  -> bf16 [B,H,D,T] (transposed so PV B-frags are contiguous)
// ---------------------------------------------------------------------------
__global__ __launch_bounds__(256)
void gemm_bf16_wmma(const bf16_t* __restrict__ A, const bf16_t* __restrict__ W,
                    const float* __restrict__ bias, void* __restrict__ out, int mode) {
  const int lane = threadIdx.x & 31;
  const int half = lane >> 4;
  const int w    = threadIdx.x >> 5;
  const int wm   = w >> 2;          // 0..1
  const int wn   = w & 3;           // 0..3
  const int mBase = blockIdx.y * 128 + wm * 64;
  const int nBase = blockIdx.x * 128 + wn * 32;

  v8f acc[4][2];
#pragma unroll
  for (int mt = 0; mt < 4; ++mt)
#pragma unroll
    for (int nt = 0; nt < 2; ++nt) acc[mt][nt] = (v8f){};

  for (int k0 = 0; k0 < EMB; k0 += 32) {
    if (k0 + 128 < EMB) { // prefetch a few k-steps ahead (global_prefetch_b8)
      __builtin_prefetch(A + (size_t)(mBase + (lane & 15)) * EMB + k0 + 128, 0, 1);
      __builtin_prefetch(W + (size_t)(nBase + (lane & 15)) * EMB + k0 + 128, 0, 1);
    }
    v16bf af[4];
#pragma unroll
    for (int mt = 0; mt < 4; ++mt)
      af[mt] = load_a_frag(A, EMB, mBase + mt * 16 + (lane & 15), k0, half);
    v16bf bf[2];
#pragma unroll
    for (int nt = 0; nt < 2; ++nt)
      bf[nt] = *(const v16bf*)(W + (size_t)(nBase + nt * 16 + (lane & 15)) * EMB +
                               k0 + (half << 4));
#pragma unroll
    for (int mt = 0; mt < 4; ++mt)
#pragma unroll
      for (int nt = 0; nt < 2; ++nt)
        acc[mt][nt] = wmma_bf16(af[mt], bf[nt], acc[mt][nt]);
  }

#pragma unroll
  for (int mt = 0; mt < 4; ++mt) {
#pragma unroll
    for (int nt = 0; nt < 2; ++nt) {
      int n = nBase + nt * 16 + (lane & 15);
      float bv = bias[n];
#pragma unroll
      for (int j = 0; j < 8; ++j) {
        int m = mBase + mt * 16 + j + (half << 3);
        float v = acc[mt][nt][j] + bv;
        if (mode == 0) {
          ((float*)out)[(size_t)m * EMB + n] = v;
        } else {
          int t = m >> 2, b = m & 3, h = n >> 6, d = n & 63;
          if (mode == 1)
            ((bf16_t*)out)[(((size_t)(b * NH + h)) * T_LEN + t) * HD + d] =
                (bf16_t)(v * 0.125f);
          else if (mode == 2)
            ((bf16_t*)out)[(((size_t)(b * NH + h)) * T_LEN + t) * HD + d] = (bf16_t)v;
          else
            ((bf16_t*)out)[(((size_t)(b * NH + h)) * HD + d) * T_LEN + t] = (bf16_t)v;
        }
      }
    }
  }
}

// ---------------------------------------------------------------------------
// Attention core: one block = (b, 32 query rows), loops over all 16 heads so
// the head-averaged probs accumulate deterministically without atomics.
// Dynamic LDS: 32 x 2048 bf16 scores/probs = 128 KB -> 2 blocks per WGP
// (320 KB LDS), and P-fragments for the PV WMMA are direct ds_loads with no
// f32->bf16 conversion in the k-loop.
// ---------------------------------------------------------------------------
__global__ __launch_bounds__(256)
void mha_core(const bf16_t* __restrict__ Qh, const bf16_t* __restrict__ Kh,
              const bf16_t* __restrict__ Vt, bf16_t* __restrict__ attn_bf,
              float* __restrict__ avg_out) {
  extern __shared__ bf16_t smem[];         // [32][2048] bf16 scores/probs
  __shared__ float rmax[32], rinv[32];

  const int lane = threadIdx.x & 31;
  const int half = lane >> 4;
  const int w    = threadIdx.x >> 5;       // 0..7
  const int t0   = blockIdx.x * 32;
  const int b    = blockIdx.y;

  for (int h = 0; h < NH; ++h) {
    const bf16_t* Qb = Qh + (size_t)(b * NH + h) * T_LEN * HD;
    const bf16_t* Kb = Kh + (size_t)(b * NH + h) * T_LEN * HD;
    const bf16_t* Vb = Vt + (size_t)(b * NH + h) * HD * T_LEN;

    // ---- scores = (Q * 1/8) @ K^T -> LDS (bf16) ------------------------
    v16bf qf[2][2];
#pragma unroll
    for (int mt = 0; mt < 2; ++mt)
#pragma unroll
      for (int ks = 0; ks < 2; ++ks)
        qf[mt][ks] = load_a_frag(Qb, HD, t0 + mt * 16 + (lane & 15), ks * 32, half);

    for (int j = 0; j < 16; ++j) {         // each wave owns 16 column tiles
      int col = w * 256 + j * 16;
      const bf16_t* kp = Kb + (size_t)(col + (lane & 15)) * HD + (half << 4);
      v16bf kf0 = *(const v16bf*)(kp);
      v16bf kf1 = *(const v16bf*)(kp + 32);
#pragma unroll
      for (int mt = 0; mt < 2; ++mt) {
        v8f s = (v8f){};
        s = wmma_bf16(qf[mt][0], kf0, s);
        s = wmma_bf16(qf[mt][1], kf1, s);
#pragma unroll
        for (int jj = 0; jj < 8; ++jj) {
          int row = mt * 16 + jj + (half << 3);
          smem[row * 2048 + col + (lane & 15)] = (bf16_t)s[jj];
        }
      }
    }
    __syncthreads();

    // ---- softmax stats: wave w -> rows 4w..4w+3 ------------------------
    for (int rr = 0; rr < 4; ++rr) {
      int row = w * 4 + rr;
      float mx = -3.0e38f;
      for (int s = lane; s < 2048; s += 32)
        mx = fmaxf(mx, (float)smem[row * 2048 + s]);
#pragma unroll
      for (int off = 16; off; off >>= 1) mx = fmaxf(mx, __shfl_xor(mx, off, 32));
      float sum = 0.f;
      for (int s = lane; s < 2048; s += 32)
        sum += __expf((float)smem[row * 2048 + s] - mx);
#pragma unroll
      for (int off = 16; off; off >>= 1) sum += __shfl_xor(sum, off, 32);
      if (lane == 0) { rmax[row] = mx; rinv[row] = 1.0f / sum; }
    }
    __syncthreads();

    // ---- normalize probs (bf16 in-place) + accumulate head-average -----
    for (int idx = threadIdx.x; idx < 32 * 2048; idx += 256) {
      int row = idx >> 11, s = idx & 2047;
      float p = __expf((float)smem[idx] - rmax[row]) * rinv[row];
      smem[idx] = (bf16_t)p;
      float* ap = avg_out + ((size_t)b * T_LEN + (t0 + row)) * T_LEN + s;
      float prev = (h == 0) ? 0.0f : *ap;      // first head initializes
      *ap = prev + p * (1.0f / NH);
    }
    __syncthreads();

    // ---- attn = P @ V : each wave owns one 16x16 tile (2 mt x 4 d) -----
    {
      int mt = w >> 2;
      int d0 = (w & 3) * 16;
      v8f acc = (v8f){};
      for (int s0 = 0; s0 < 2048; s0 += 32) {
        int row = mt * 16 + (lane & 15);
        v16bf af = load_a_frag_lds(smem + row * 2048 + s0 + (half << 4));
        v16bf vf = *(const v16bf*)(Vb + (size_t)(d0 + (lane & 15)) * T_LEN +
                                   s0 + (half << 4));
        acc = wmma_bf16(af, vf, acc);
      }
      int d = d0 + (lane & 15);
#pragma unroll
      for (int jj = 0; jj < 8; ++jj) {
        int t = t0 + mt * 16 + jj + (half << 3);
        attn_bf[((size_t)t * BSZ + b) * EMB + h * HD + d] = (bf16_t)acc[jj];
      }
    }
    __syncthreads();   // LDS reused next head
  }
}

// ---------------------------------------------------------------------------
extern "C" void kernel_launch(void* const* d_in, const int* in_sizes, int n_in,
                              void* d_out, int out_size, void* d_ws, size_t ws_size,
                              hipStream_t stream) {
  const float* q  = (const float*)d_in[0];
  const float* k  = (const float*)d_in[1];
  const float* v  = (const float*)d_in[2];
  const float* Wq = (const float*)d_in[3];
  const float* bq = (const float*)d_in[4];
  const float* Wk = (const float*)d_in[5];
  const float* bk = (const float*)d_in[6];
  const float* Wv = (const float*)d_in[7];
  const float* bv = (const float*)d_in[8];
  const float* Wo = (const float*)d_in[9];
  const float* bo = (const float*)d_in[10];

  const size_t ACT = (size_t)MROWS * EMB;   // 8,388,608
  const size_t WEL = (size_t)EMB * EMB;     // 1,048,576

  bf16_t* base = (bf16_t*)d_ws;
  bf16_t* Xq  = base;
  bf16_t* Xk  = Xq + ACT;
  bf16_t* Xv  = Xk + ACT;
  bf16_t* WqB = Xv + ACT;
  bf16_t* WkB = WqB + WEL;
  bf16_t* WvB = WkB + WEL;
  bf16_t* WoB = WvB + WEL;
  bf16_t* Qh  = WoB + WEL;                  // [B,H,T,D]
  bf16_t* Kh  = Qh + ACT;                   // [B,H,T,D]
  bf16_t* Vt  = Kh + ACT;                   // [B,H,D,T]
  bf16_t* Abf = Vt + ACT;                   // attn as [T,B,E] bf16

  float* out_attn = (float*)d_out;                  // [T,B,E]
  float* out_avg  = out_attn + ACT;                 // [B,T,S]

  // 1) f32 -> bf16 conversions
  {
    int n4a = (int)(ACT / 4), n4w = (int)(WEL / 4);
    dim3 blk(256);
    cvt_f32_to_bf16<<<dim3((n4a + 255) / 256), blk, 0, stream>>>(q,  Xq,  n4a);
    cvt_f32_to_bf16<<<dim3((n4a + 255) / 256), blk, 0, stream>>>(k,  Xk,  n4a);
    cvt_f32_to_bf16<<<dim3((n4a + 255) / 256), blk, 0, stream>>>(v,  Xv,  n4a);
    cvt_f32_to_bf16<<<dim3((n4w + 255) / 256), blk, 0, stream>>>(Wq, WqB, n4w);
    cvt_f32_to_bf16<<<dim3((n4w + 255) / 256), blk, 0, stream>>>(Wk, WkB, n4w);
    cvt_f32_to_bf16<<<dim3((n4w + 255) / 256), blk, 0, stream>>>(Wv, WvB, n4w);
    cvt_f32_to_bf16<<<dim3((n4w + 255) / 256), blk, 0, stream>>>(Wo, WoB, n4w);
  }

  // 2) in-projections (WMMA GEMMs) with layout-transforming epilogues
  dim3 gGrid(EMB / 128, MROWS / 128);       // (8, 64)
  dim3 gBlk(256);
  gemm_bf16_wmma<<<gGrid, gBlk, 0, stream>>>(Xq, WqB, bq, Qh, 1);
  gemm_bf16_wmma<<<gGrid, gBlk, 0, stream>>>(Xk, WkB, bk, Kh, 2);
  gemm_bf16_wmma<<<gGrid, gBlk, 0, stream>>>(Xv, WvB, bv, Vt, 3);

  // 3) attention core (softmax + both matmuls + averaged probs)
  dim3 aGrid(T_LEN / 32, BSZ);              // (64, 4)
  mha_core<<<aGrid, dim3(256), 32 * 2048 * sizeof(bf16_t), stream>>>(
      Qh, Kh, Vt, Abf, out_avg);

  // 4) out-projection -> d_out [T,B,E] f32
  gemm_bf16_wmma<<<gGrid, gBlk, 0, stream>>>(Abf, WoB, bo, out_attn, 0);
}
